// SelfAttention_61727269978623
// MI455X (gfx1250) — matrix operand order repaired and checked by reading, not compiled, not run on hardware
//
#include <hip/hip_runtime.h>

// ---------------- problem constants ----------------
#define SQd     1024
#define Bd      2
#define Hd      4096
#define NHd     32
#define HDd     128
#define MEMd    1024
#define KVd     (MEMd + SQd)      // 2048
#define MROWS   (SQd * Bd)        // 2048
#define LN10000 9.210340371976184f
#define QSCALE  0.0883883476483184f   /* 1/sqrt(128); folds sqrt(HD)*coeff then *coeff */
#define MASKVAL (-40000.0f)           /* -10000 * coeff(=4) */

typedef __attribute__((ext_vector_type(16))) __bf16 v16bf;
typedef __attribute__((ext_vector_type(8)))  __bf16 v8bf;
typedef __attribute__((ext_vector_type(8)))  float  v8f;
typedef int vint4 __attribute__((vector_size(16)));

// pack two f32 -> two bf16 in a dword
__device__ __forceinline__ unsigned pk2(float a, float b) {
  union { __bf16 h[2]; unsigned u; } p;
  p.h[0] = (__bf16)a; p.h[1] = (__bf16)b;
  return p.u;
}

// ---------------- CDNA5 async global->LDS copy (ASYNCcnt) ----------------
#if __has_builtin(__builtin_amdgcn_global_load_async_to_lds_b128)
#define HAVE_ASYNC 1
__device__ __forceinline__ void cp128(const __bf16* g, __bf16* l) {
  __builtin_amdgcn_global_load_async_to_lds_b128(
      (__attribute__((address_space(1))) vint4*)(g),
      (__attribute__((address_space(3))) vint4*)(l), 0, 0);
}
#else
#define HAVE_ASYNC 0
__device__ __forceinline__ void cp128(const __bf16* g, __bf16* l) {
  *(uint4*)l = *(const uint4*)g;
}
#endif

#if HAVE_ASYNC
#  if __has_builtin(__builtin_amdgcn_s_wait_asynccnt)
#    define WAITA(n) __builtin_amdgcn_s_wait_asynccnt(n)
#  else
#    define WAITA(n) asm volatile("s_wait_asynccnt " #n ::: "memory")
#  endif
#else
#  define WAITA(n) ((void)0)
#endif

// A-operand fragment (16 x K=32, bf16) from a row-major bf16 tile.
// lane: m = lane&15, kh = lane>>4 ; VGPR0-3 hold K = kh*8+[0..7], VGPR4-7 hold K = 16+kh*8+[0..7]
__device__ __forceinline__ v16bf frag_a(const __bf16* base, int m0, int k0, int ld) {
  const int lane = threadIdx.x & 31;
  const __bf16* p = base + (m0 + (lane & 15)) * ld + k0 + ((lane >> 4) << 3);
  union { v16bf v; v8bf h[2]; } u;
  u.h[0] = *(const v8bf*)(p);
  u.h[1] = *(const v8bf*)(p + 16);
  return u.v;
}

// B-operand fragment (K=32 x 16, bf16): tile row n holds B-matrix column n (B^T row-major).
// lane: n = lane&15, kh = lane>>4 ; VGPR0-3 hold K = kh*8+[0..7], VGPR4-7 hold K = 16+kh*8+[0..7]
__device__ __forceinline__ v16bf frag_b(const __bf16* base, int n0, int k0, int ld) {
  const int lane = threadIdx.x & 31;
  const __bf16* p = base + (n0 + (lane & 15)) * ld + k0 + ((lane >> 4) << 3);
  union { v16bf v; v8bf h[2]; } u;
  u.h[0] = *(const v8bf*)(p);
  u.h[1] = *(const v8bf*)(p + 16);
  return u.v;
}

__device__ __forceinline__ v8f wmma_bf16(v16bf a, v16bf b, v8f c) {
  return __builtin_amdgcn_wmma_f32_16x16x32_bf16(false, a, false, b, (short)0, c, false, false);
}

// stage a 128x64 A tile and a 128x64 B tile (bf16, row-major stride Hd) into LDS
__device__ __forceinline__ void stage_ab(const __bf16* Ag, const __bf16* Bg,
                                         __bf16* sA, __bf16* sB, int kt, int tid) {
#pragma unroll
  for (int j = 0; j < 4; ++j) {
    const int ch = j * 256 + tid;       // 1024 chunks of 8 bf16 per tile
    const int r  = ch >> 3;
    const int c8 = (ch & 7) * 8;
    cp128(Ag + r * Hd + kt + c8, sA + r * 64 + c8);
    cp128(Bg + r * Hd + kt + c8, sB + r * 64 + c8);
  }
}

// ============================================================================
// Kernel 0: generic f32 -> bf16 conversion (one-time; makes weights L2-resident
// bf16 so GEMM staging is pure b128 copies eligible for async-to-LDS)
// ============================================================================
__global__ __launch_bounds__(256) void k_cvt(const float* __restrict__ src,
                                             unsigned short* __restrict__ dst_) {
  __bf16* dst = (__bf16*)dst_;
  const int i4 = (blockIdx.x * 256 + threadIdx.x) * 4;
  const float4 x = *(const float4*)(src + i4);
  *(uint2*)(dst + i4) = make_uint2(pk2(x.x, x.y), pk2(x.z, x.w));
}

// ============================================================================
// Kernel 2: convert mem (B, MEM, 2, NH, HD) f32 -> bf16 K/V caches (B,NH,KV,HD)
// ============================================================================
__global__ __launch_bounds__(256) void k_memcvt(const float* __restrict__ mem,
                                                unsigned short* __restrict__ kbuf_,
                                                unsigned short* __restrict__ vbuf_) {
  __bf16* kbuf = (__bf16*)kbuf_;
  __bf16* vbuf = (__bf16*)vbuf_;
  const int idx4 = (blockIdx.x * 256 + threadIdx.x) * 4;
  const int c  = idx4 & (HDd - 1);
  int t1 = idx4 >> 7;
  const int h  = t1 & (NHd - 1);
  t1 >>= 5;
  const int kv = t1 & 1;
  t1 >>= 1;
  const int t  = t1 & (MEMd - 1);
  const int b  = t1 >> 10;
  const float4 x = *(const float4*)(mem + idx4);
  __bf16* dst = (kv == 0 ? kbuf : vbuf) + ((b * NHd + h) * KVd + t) * HDd + c;
  *(uint2*)dst = make_uint2(pk2(x.x, x.y), pk2(x.z, x.w));
}

// ============================================================================
// Kernel 1: QKV GEMM (2048 x 12288 x 4096) + bias + RoPE fused epilogue.
// CTA tile 128x128, BK=64, async double-buffered LDS. 8 waves x (16 rows x 128
// cols) so RoPE pairs (c, c+64) are accumulators nf and nf+4 of the same lane.
// B-fragments preloaded into an array so LDS loads overlap the matrix pipe.
// ============================================================================
__global__ __launch_bounds__(256) void k_qkv_rope(const unsigned short* __restrict__ A_,  // hidden bf16
                                                  const unsigned short* __restrict__ W_,  // qkv_w bf16
                                                  const float* __restrict__ bias,
                                                  unsigned short* __restrict__ qbuf_,
                                                  unsigned short* __restrict__ kbuf_,
                                                  unsigned short* __restrict__ vbuf_) {
  const __bf16* Ab = (const __bf16*)A_;
  const __bf16* Wb = (const __bf16*)W_;
  __bf16* qbuf = (__bf16*)qbuf_;
  __bf16* kbuf = (__bf16*)kbuf_;
  __bf16* vbuf = (__bf16*)vbuf_;
  __shared__ __bf16 sA[2][128 * 64];
  __shared__ __bf16 sB[2][128 * 64];

  const int tid  = threadIdx.x;
  const int wave = tid >> 5;
  const int lane = tid & 31;
  const int nn   = lane & 15;
  const int half = lane >> 4;
  const int m0   = blockIdx.x * 128;
  const int n0   = blockIdx.y * 128;
  const __bf16* Ag = Ab + m0 * Hd;
  const __bf16* Wg = Wb + n0 * Hd;

  v8f acc[8] = {};

  stage_ab(Ag, Wg, sA[0], sB[0], 0, tid);
  stage_ab(Ag, Wg, sA[1], sB[1], 64, tid);
  int pb = 0;
  for (int kt = 0; kt < Hd - 64; kt += 64, pb ^= 1) {
    WAITA(8);                // current buffer's 8 async ops complete (in-order)
    __syncthreads();
#pragma unroll
    for (int ks = 0; ks < 64; ks += 32) {
      const v16bf af = frag_a(sA[pb], wave * 16, ks, 64);
      v16bf bfr[8];
#pragma unroll
      for (int nf = 0; nf < 8; ++nf)
        bfr[nf] = frag_b(sB[pb], nf * 16, ks, 64);
#pragma unroll
      for (int nf = 0; nf < 8; ++nf)
        acc[nf] = wmma_bf16(af, bfr[nf], acc[nf]);
    }
    __syncthreads();
    if (kt + 128 < Hd) stage_ab(Ag, Wg, sA[pb], sB[pb], kt + 128, tid);
  }
  WAITA(0);
  __syncthreads();
#pragma unroll
  for (int ks = 0; ks < 64; ks += 32) {
    const v16bf af = frag_a(sA[pb], wave * 16, ks, 64);
    v16bf bfr[8];
#pragma unroll
    for (int nf = 0; nf < 8; ++nf)
      bfr[nf] = frag_b(sB[pb], nf * 16, ks, 64);
#pragma unroll
    for (int nf = 0; nf < 8; ++nf)
      acc[nf] = wmma_bf16(af, bfr[nf], acc[nf]);
  }

  // epilogue: bias + RoPE + scatter into attention-layout bf16 buffers
  const int sect = n0 % 384;      // 0 -> q, 128 -> k, 256 -> v
  const int head = n0 / 384;
#pragma unroll
  for (int r = 0; r < 8; ++r) {
    const int m = m0 + wave * 16 + 8 * half + r;
    const int s = m >> 1;         // B=2
    const int b = m & 1;
    const float pos = (float)(MEMd + s);
#pragma unroll
    for (int nf = 0; nf < 4; ++nf) {
      const int c1 = nf * 16 + nn;      // 0..63
      const int c2 = c1 + 64;
      float x1 = acc[nf][r]     + bias[n0 + c1];
      float x2 = acc[nf + 4][r] + bias[n0 + c2];
      if (sect == 256) {                 // V: bias only
        const int vi = ((b * NHd + head) * KVd + MEMd + s) * HDd;
        vbuf[vi + c1] = (__bf16)x1;
        vbuf[vi + c2] = (__bf16)x2;
      } else {                           // Q or K: rotary
        const float invf = __expf(-(float)c1 * (LN10000 / 64.0f));
        float sn, cs;
        __sincosf(pos * invf, &sn, &cs);
        float y1 = x1 * cs - x2 * sn;
        float y2 = x2 * cs + x1 * sn;
        if (sect == 0) {                 // Q: fold 1/sqrt(HD)
          y1 *= QSCALE; y2 *= QSCALE;
          const int qi = ((b * NHd + head) * SQd + s) * HDd;
          qbuf[qi + c1] = (__bf16)y1;
          qbuf[qi + c2] = (__bf16)y2;
        } else {                         // K
          const int ki = ((b * NHd + head) * KVd + MEMd + s) * HDd;
          kbuf[ki + c1] = (__bf16)y1;
          kbuf[ki + c2] = (__bf16)y2;
        }
      }
    }
  }
}

// ============================================================================
// Kernel 3: flash attention per (b, head). CTA: 128 q rows, 8 waves x 16 rows.
// ============================================================================
__global__ __launch_bounds__(256) void k_attn(const unsigned short* __restrict__ qbuf_,
                                              const unsigned short* __restrict__ kbuf_,
                                              const unsigned short* __restrict__ vbuf_,
                                              unsigned short* __restrict__ ctx_) {
  const __bf16* qbuf = (const __bf16*)qbuf_;
  const __bf16* kbuf = (const __bf16*)kbuf_;
  const __bf16* vbuf = (const __bf16*)vbuf_;
  __bf16* ctx = (__bf16*)ctx_;

  __shared__ __bf16 sK[64 * 128];     // K tile, also Q staging
  __shared__ __bf16 sVT[128 * 64];    // V tile transposed (HD x kv)
  __shared__ __bf16 sP[8 * 16 * 64];  // per-wave P tiles

  const int tid  = threadIdx.x;
  const int wave = tid >> 5;
  const int lane = tid & 31;
  const int nn   = lane & 15;
  const int half = lane >> 4;
  const int s0   = blockIdx.x * 128;
  const int bh   = blockIdx.y;              // b*NH + head
  const int b    = bh >> 5;
  const int head = bh & 31;

  const __bf16* qg = qbuf + (bh * SQd + s0) * HDd;
  const __bf16* kg = kbuf + bh * KVd * HDd;
  const __bf16* vg = vbuf + bh * KVd * HDd;

  // ---- load Q fragments to registers (two 64-row halves staged via sK) ----
  v16bf qf[4] = {};
  for (int hh = 0; hh < 2; ++hh) {
#pragma unroll
    for (int j = 0; j < 4; ++j) {
      const int ch = j * 256 + tid;
      const int r  = ch >> 4;
      const int c8 = (ch & 15) * 8;
      *(uint4*)(sK + r * 128 + c8) = *(const uint4*)(qg + (hh * 64 + r) * HDd + c8);
    }
    __syncthreads();
    if ((wave >> 2) == hh) {
#pragma unroll
      for (int ks = 0; ks < 4; ++ks)
        qf[ks] = frag_a(sK, (wave & 3) * 16, ks * 32, 128);
    }
    __syncthreads();
  }

  v8f o[8] = {};
  float mrun[8], lrun[8];
#pragma unroll
  for (int r = 0; r < 8; ++r) { mrun[r] = -3.0e38f; lrun[r] = 0.0f; }

  const int nkv = s0 + 128 + MEMd;           // causal horizon (multiple of 64)
  for (int kt = 0; kt < nkv; kt += 64) {
    // ---- K tile (64x128): straight b128 copies ----
#pragma unroll
    for (int j = 0; j < 4; ++j) {
      const int ch = j * 256 + tid;
      const int r  = ch >> 4;
      const int c8 = (ch & 15) * 8;
      *(uint4*)(sK + r * 128 + c8) = *(const uint4*)(kg + (kt + r) * HDd + c8);
    }
    // ---- V^T tile: 4 rows x 8 cols per thread, packed ds_store_b64 ----
    {
      const int t0 = (tid >> 4) * 4;       // 0..60
      const int c0 = (tid & 15) * 8;       // 0..120
      const v8bf r0v = *(const v8bf*)(vg + (kt + t0 + 0) * HDd + c0);
      const v8bf r1v = *(const v8bf*)(vg + (kt + t0 + 1) * HDd + c0);
      const v8bf r2v = *(const v8bf*)(vg + (kt + t0 + 2) * HDd + c0);
      const v8bf r3v = *(const v8bf*)(vg + (kt + t0 + 3) * HDd + c0);
#pragma unroll
      for (int e = 0; e < 8; ++e) {
        union { __bf16 h[4]; uint2 u; } q;
        q.h[0] = r0v[e]; q.h[1] = r1v[e]; q.h[2] = r2v[e]; q.h[3] = r3v[e];
        *(uint2*)(sVT + (c0 + e) * 64 + t0) = q.u;
      }
    }
    __syncthreads();

    // ---- S = Q K^T (16 WMMAs, K-frags preloaded per k-step) ----
    v8f sfr[4] = {};
#pragma unroll
    for (int ks = 0; ks < 4; ++ks) {
      v16bf kfr[4];
#pragma unroll
      for (int nf = 0; nf < 4; ++nf)
        kfr[nf] = frag_b(sK, nf * 16, ks * 32, 128);
#pragma unroll
      for (int nf = 0; nf < 4; ++nf)
        sfr[nf] = wmma_bf16(qf[ks], kfr[nf], sfr[nf]);
    }

    // ---- mask + online softmax ----
    float sv[4][8];
#pragma unroll
    for (int nf = 0; nf < 4; ++nf)
#pragma unroll
      for (int r = 0; r < 8; ++r) {
        const int t = kt + nf * 16 + nn;
        const int s = s0 + wave * 16 + 8 * half + r;
        sv[nf][r] = (t - MEMd > s) ? MASKVAL : sfr[nf][r];
      }
#pragma unroll
    for (int r = 0; r < 8; ++r) {
      float mx = fmaxf(fmaxf(sv[0][r], sv[1][r]), fmaxf(sv[2][r], sv[3][r]));
#pragma unroll
      for (int off = 1; off < 16; off <<= 1)
        mx = fmaxf(mx, __shfl_xor(mx, off, 32));
      const float mnew = fmaxf(mrun[r], mx);
      const float corr = __expf(mrun[r] - mnew);
      mrun[r] = mnew;
      float rs = 0.0f;
#pragma unroll
      for (int nf = 0; nf < 4; ++nf) {
        const float p = __expf(sv[nf][r] - mnew);
        sv[nf][r] = p;
        rs += p;
      }
#pragma unroll
      for (int off = 1; off < 16; off <<= 1)
        rs += __shfl_xor(rs, off, 32);
      lrun[r] = lrun[r] * corr + rs;
#pragma unroll
      for (int nf = 0; nf < 8; ++nf)
        o[nf][r] *= corr;
    }

    // ---- P to wave-private LDS (C-layout -> A-layout) ----
    __bf16* myP = sP + wave * (16 * 64);
#pragma unroll
    for (int nf = 0; nf < 4; ++nf)
#pragma unroll
      for (int r = 0; r < 8; ++r)
        myP[(8 * half + r) * 64 + nf * 16 + nn] = (__bf16)sv[nf][r];

    // ---- O += P V (16 WMMAs, V-frags preloaded in groups of 4) ----
#pragma unroll
    for (int ks = 0; ks < 2; ++ks) {
      const v16bf pf = frag_a(myP, 0, ks * 32, 64);
#pragma unroll
      for (int g = 0; g < 2; ++g) {
        v16bf vfr[4];
#pragma unroll
        for (int j = 0; j < 4; ++j)
          vfr[j] = frag_b(sVT, (g * 4 + j) * 16, ks * 32, 64);
#pragma unroll
        for (int j = 0; j < 4; ++j)
          o[g * 4 + j] = wmma_bf16(pf, vfr[j], o[g * 4 + j]);
      }
    }
    __syncthreads();
  }

  // ---- write ctx (bf16, row-major (s*B+b, H)) ----
#pragma unroll
  for (int r = 0; r < 8; ++r) {
    const int s = s0 + wave * 16 + 8 * half + r;
    const float inv = 1.0f / lrun[r];
    const int rowbase = (s * Bd + b) * Hd + head * HDd;
#pragma unroll
    for (int nf = 0; nf < 8; ++nf)
      ctx[rowbase + nf * 16 + nn] = (__bf16)(o[nf][r] * inv);
  }
}

// ============================================================================
// Kernel 4: dense GEMM (2048 x 4096 x 4096) + bias, f32 output.
// Wave tile 32x64 (2 A-frags x 4 B-frags): 12 ds_load_b128 per 8 WMMAs.
// ============================================================================
__global__ __launch_bounds__(256) void k_dense(const unsigned short* __restrict__ ctx_,
                                               const unsigned short* __restrict__ W_,
                                               const float* __restrict__ bias,
                                               float* __restrict__ out) {
  const __bf16* ctx = (const __bf16*)ctx_;
  const __bf16* Wb  = (const __bf16*)W_;
  __shared__ __bf16 sA[2][128 * 64];
  __shared__ __bf16 sB[2][128 * 64];

  const int tid  = threadIdx.x;
  const int wave = tid >> 5;
  const int lane = tid & 31;
  const int nn   = lane & 15;
  const int half = lane >> 4;
  const int wm   = wave >> 1;          // 0..3 : 32-row strip
  const int wn   = wave & 1;           // 0..1 : 64-col strip
  const int m0   = blockIdx.x * 128;
  const int n0   = blockIdx.y * 128;
  const __bf16* Ag = ctx + m0 * Hd;
  const __bf16* Wg = Wb + n0 * Hd;

  v8f acc[2][4] = {};

  stage_ab(Ag, Wg, sA[0], sB[0], 0, tid);
  stage_ab(Ag, Wg, sA[1], sB[1], 64, tid);
  int pb = 0;
  for (int kt = 0; kt < Hd - 64; kt += 64, pb ^= 1) {
    WAITA(8);
    __syncthreads();
#pragma unroll
    for (int ks = 0; ks < 64; ks += 32) {
      const v16bf af0 = frag_a(sA[pb], wm * 32, ks, 64);
      const v16bf af1 = frag_a(sA[pb], wm * 32 + 16, ks, 64);
      v16bf bfr[4];
#pragma unroll
      for (int nf = 0; nf < 4; ++nf)
        bfr[nf] = frag_b(sB[pb], wn * 64 + nf * 16, ks, 64);
#pragma unroll
      for (int nf = 0; nf < 4; ++nf) {
        acc[0][nf] = wmma_bf16(af0, bfr[nf], acc[0][nf]);
        acc[1][nf] = wmma_bf16(af1, bfr[nf], acc[1][nf]);
      }
    }
    __syncthreads();
    if (kt + 128 < Hd) stage_ab(Ag, Wg, sA[pb], sB[pb], kt + 128, tid);
  }
  WAITA(0);
  __syncthreads();
#pragma unroll
  for (int ks = 0; ks < 64; ks += 32) {
    const v16bf af0 = frag_a(sA[pb], wm * 32, ks, 64);
    const v16bf af1 = frag_a(sA[pb], wm * 32 + 16, ks, 64);
    v16bf bfr[4];
#pragma unroll
    for (int nf = 0; nf < 4; ++nf)
      bfr[nf] = frag_b(sB[pb], wn * 64 + nf * 16, ks, 64);
#pragma unroll
    for (int nf = 0; nf < 4; ++nf) {
      acc[0][nf] = wmma_bf16(af0, bfr[nf], acc[0][nf]);
      acc[1][nf] = wmma_bf16(af1, bfr[nf], acc[1][nf]);
    }
  }

#pragma unroll
  for (int mf = 0; mf < 2; ++mf)
#pragma unroll
    for (int r = 0; r < 8; ++r) {
      const int m = m0 + wm * 32 + mf * 16 + 8 * half + r;
#pragma unroll
      for (int nf = 0; nf < 4; ++nf) {
        const int oc = n0 + wn * 64 + nf * 16 + nn;
        out[m * Hd + oc] = acc[mf][nf][r] + bias[oc];
      }
    }
}

// ============================================================================
extern "C" void kernel_launch(void* const* d_in, const int* in_sizes, int n_in,
                              void* d_out, int out_size, void* d_ws, size_t ws_size,
                              hipStream_t stream) {
  const float* hidden  = (const float*)d_in[0];
  // d_in[1] = mask (unused, derived analytically), d_in[2] = position_ids (unused)
  const float* mem     = (const float*)d_in[3];
  const float* qkv_w   = (const float*)d_in[4];
  const float* qkv_b   = (const float*)d_in[5];
  const float* dense_w = (const float*)d_in[6];
  const float* dense_b = (const float*)d_in[7];
  float* out = (float*)d_out;

  // workspace (bf16 as ushort): q | k | v | ctx | a(hidden) | wqkv | wdense  ~= 240 MB
  unsigned short* qbuf = (unsigned short*)d_ws;
  unsigned short* kbuf = qbuf + (size_t)Bd * NHd * SQd * HDd;   //  8,388,608
  unsigned short* vbuf = kbuf + (size_t)Bd * NHd * KVd * HDd;   // 16,777,216
  unsigned short* ctx  = vbuf + (size_t)Bd * NHd * KVd * HDd;   // 16,777,216
  unsigned short* abuf = ctx  + (size_t)MROWS * Hd;             //  8,388,608
  unsigned short* wq   = abuf + (size_t)MROWS * Hd;             //  8,388,608
  unsigned short* wd   = wq   + (size_t)3 * Hd * Hd;            // 50,331,648

  k_cvt<<<dim3((MROWS * Hd) / 1024), 256, 0, stream>>>(hidden, abuf);
  k_cvt<<<dim3((3 * Hd * Hd) / 1024), 256, 0, stream>>>(qkv_w, wq);
  k_cvt<<<dim3((Hd * Hd) / 1024), 256, 0, stream>>>(dense_w, wd);
  k_memcvt<<<dim3(16384), 256, 0, stream>>>(mem, kbuf, vbuf);
  k_qkv_rope<<<dim3(MROWS / 128, (3 * Hd) / 128), 256, 0, stream>>>(abuf, wq, qkv_b,
                                                                    qbuf, kbuf, vbuf);
  k_attn<<<dim3(SQd / 128, Bd * NHd), 256, 0, stream>>>(qbuf, kbuf, vbuf, ctx);
  k_dense<<<dim3(MROWS / 128, Hd / 128), 256, 0, stream>>>(ctx, wd, dense_b, out);
}
